// TFMamba_46127948759822
// MI455X (gfx1250) — compile-verified, matrix operand order checked by reading, not run on hardware
//
#include <hip/hip_runtime.h>
#include <hip/hip_bf16.h>
#include <math.h>

typedef _Float16 v16h __attribute__((ext_vector_type(16)));
typedef _Float16 half8 __attribute__((ext_vector_type(8)));
typedef float    v8f  __attribute__((ext_vector_type(8)));

#define BQ 8
#define TT 12
#define NN 883
#define D_MODEL 152
#define D2 304
#define SSN 128
#define NLAYERS 3
#define BN_ROWS (BQ * NN)            // 7064
#define M_ROWS  (BN_ROWS * TT)       // 84768
#define DMP 160                      // padded 152
#define D2P 320                      // padded 304
#define FUSEW 576                    // fc1(320) ++ fc2(128) ++ fc3(128)

// ---------------------------------------------------------------------------
// Weight pad + fp32->fp16, stored TRANSPOSED: dst[(col0+n)*Kp + k]
// (so GEMM B staging is a contiguous 16B load along k)
// ---------------------------------------------------------------------------
__global__ void pad_wT_f16(const float* __restrict__ src, _Float16* __restrict__ dst,
                           int K, int Nn, int Kp, int Nseg, int col0) {
    int idx = blockIdx.x * 256 + threadIdx.x;
    if (idx >= Nseg * Kp) return;
    int k = idx % Kp, n = idx / Kp;
    float v = (k < K && n < Nn) ? src[(size_t)k * Nn + n] : 0.f;
    dst[(size_t)(col0 + n) * Kp + k] = (_Float16)v;
}

__global__ void pad_b_f32(const float* __restrict__ src, float* __restrict__ dst,
                          int Nn, int Nseg, int col0) {
    int n = blockIdx.x * 256 + threadIdx.x;
    if (n >= Nseg) return;
    dst[col0 + n] = (n < Nn) ? src[n] : 0.f;
}

// ---------------------------------------------------------------------------
// Embedding: build h (M_ROWS x DMP) fp32, zero-padded
// ---------------------------------------------------------------------------
__global__ void embed_kernel(const float* __restrict__ x,
                             const float* __restrict__ W_in, const float* __restrict__ b_in,
                             const float* __restrict__ tod, const float* __restrict__ dow,
                             const float* __restrict__ adp, float* __restrict__ h) {
    int idx = blockIdx.x * 256 + threadIdx.x;
    if (idx >= M_ROWS * DMP) return;
    int c  = idx % DMP;
    int rt = idx / DMP;        // row = bn*T + t
    int t  = rt % TT;
    int bn = rt / TT;
    int b  = bn / NN;
    int n  = bn % NN;
    const float* xe = x + (((size_t)b * TT + t) * NN + n) * 3;
    float v = 0.f;
    if (c < 24) {
        v = b_in[c] + xe[0] * W_in[c] + xe[1] * W_in[24 + c] + xe[2] * W_in[48 + c];
    } else if (c < 48) {
        int ti = (int)(xe[1] * 288.f);
        ti = ti < 0 ? 0 : (ti > 287 ? 287 : ti);
        v = tod[ti * 24 + (c - 24)];
    } else if (c < 72) {
        int di = (int)xe[2];
        di = di < 0 ? 0 : (di > 6 ? 6 : di);
        v = dow[di * 24 + (c - 48)];
    } else if (c < 152) {
        v = adp[((size_t)t * NN + n) * 80 + (c - 72)];
    }
    h[idx] = v;
}

// ---------------------------------------------------------------------------
// RMSNorm: wave-per-row, fp32 in -> fp16 out (padded cols -> 0)
// ---------------------------------------------------------------------------
__global__ __launch_bounds__(256)
void rmsnorm_kernel(const float* __restrict__ h, const float* __restrict__ norm_w,
                    _Float16* __restrict__ xn) {
    int lane = threadIdx.x & 31, wid = threadIdx.x >> 5;
    int row = blockIdx.x * 8 + wid;
    if (row >= M_ROWS) return;
    const float* hr = h + (size_t)row * DMP;
    float s = 0.f;
    for (int c = lane; c < DMP; c += 32) { float v = hr[c]; s += v * v; }  // pad cols are 0
    for (int off = 16; off; off >>= 1) s += __shfl_xor(s, off, 32);
    float scale = rsqrtf(s / 152.0f + 1e-5f);
    _Float16* xr = xn + (size_t)row * DMP;
    for (int c = lane; c < DMP; c += 32) {
        float v = (c < 152) ? hr[c] * scale * norm_w[c] : 0.f;
        xr[c] = (_Float16)v;
    }
}

// ---------------------------------------------------------------------------
// WMMA GEMM: C[M x Nstore] = A[M x Kp](f16) * B[Kp x N](f16, stored B^T) + bias
// 256 threads = 8 waves; block tile 128x64; wave tile 32x32 (4 accumulators).
// spCols: softplus applied to columns < spCols. EPI: 0 = f16 out, 1 = f32 out.
// ---------------------------------------------------------------------------
template <int EPI>
__global__ __launch_bounds__(256)
void gemm_wmma(const _Float16* __restrict__ A, int lda,
               const _Float16* __restrict__ Bt, int ldbt /* == Kp */,
               const float* __restrict__ bias,
               void* __restrict__ Cout, int ldc, int Nstore, int Kp, int spCols) {
    __shared__ __align__(16) _Float16 Alds[128][40];  // padded rows (80B, 16B aligned)
    __shared__ __align__(16) _Float16 Blds[64][40];   // B^T: [n][k]

    const int tid  = threadIdx.x;
    const int lane = tid & 31, wid = tid >> 5;
    const int wm = wid & 3;          // 4 M-slots of 32 rows
    const int wn = wid >> 2;         // 2 N-slots of 32 cols
    const int rowBase = blockIdx.x * 128;
    const int colBase = blockIdx.y * 64;

    v8f acc[2][2];
#pragma unroll
    for (int r = 0; r < 2; ++r)
#pragma unroll
        for (int c = 0; c < 2; ++c) acc[r][c] = (v8f){0.f,0.f,0.f,0.f,0.f,0.f,0.f,0.f};

    const int arow = tid >> 1;           // 0..127
    const int acol = (tid & 1) * 16;     // 0 or 16
    const int bn   = tid >> 2;           // 0..63
    const int bk8  = (tid & 3) * 8;      // 0,8,16,24

    const int half_ = lane >> 4;
    const int mrow  = lane & 15;

    for (int kk = 0; kk < Kp; kk += 32) {
        // ---- stage A tile (128 x 32): two 16B vector loads/stores per thread
        half8 av0 = {}, av1 = {};
        int gr = rowBase + arow;
        if (gr < M_ROWS) {
            const _Float16* ap = A + (size_t)gr * lda + kk + acol;
            av0 = *(const half8*)ap;
            av1 = *(const half8*)(ap + 8);
            if (kk + 32 < Kp) __builtin_prefetch(ap + 32, 0, 1);
        }
        *(half8*)(&Alds[arow][acol])     = av0;
        *(half8*)(&Alds[arow][acol + 8]) = av1;

        // ---- stage B^T tile (64 n x 32 k): one contiguous 16B load per thread
        half8 bv = *(const half8*)(Bt + (size_t)(colBase + bn) * ldbt + kk + bk8);
        *(half8*)(&Blds[bn][bk8]) = bv;
        __syncthreads();

        // ---- build fragments (ISA 16-bit A-matrix 16x32 layout)
        v16h af[2], bf[2];
#pragma unroll
        for (int r = 0; r < 2; ++r) {
            half8 lo = *(const half8*)(&Alds[wm * 32 + r * 16 + mrow][half_ * 8]);
            half8 hi = *(const half8*)(&Alds[wm * 32 + r * 16 + mrow][16 + half_ * 8]);
#pragma unroll
            for (int i = 0; i < 8; ++i) { af[r][i] = lo[i]; af[r][8 + i] = hi[i]; }
        }
#pragma unroll
        for (int c = 0; c < 2; ++c) {
            half8 lo = *(const half8*)(&Blds[wn * 32 + c * 16 + mrow][half_ * 8]);
            half8 hi = *(const half8*)(&Blds[wn * 32 + c * 16 + mrow][16 + half_ * 8]);
#pragma unroll
            for (int i = 0; i < 8; ++i) { bf[c][i] = lo[i]; bf[c][8 + i] = hi[i]; }
        }

#pragma unroll
        for (int r = 0; r < 2; ++r)
#pragma unroll
            for (int c = 0; c < 2; ++c)
                acc[r][c] = __builtin_amdgcn_wmma_f32_16x16x32_f16(
                    false, af[r], false, bf[c], (short)0, acc[r][c], false, false);
        __syncthreads();
    }

    // ---- epilogue
#pragma unroll
    for (int c = 0; c < 2; ++c) {
        int col = colBase + wn * 32 + c * 16 + mrow;
        if (col >= Nstore) continue;
        float bvv = bias[col];
        bool sp = col < spCols;
#pragma unroll
        for (int r = 0; r < 2; ++r) {
#pragma unroll
            for (int v = 0; v < 8; ++v) {
                int row = rowBase + wm * 32 + r * 16 + half_ * 8 + v;
                if (row >= M_ROWS) continue;
                float val = acc[r][c][v] + bvv;
                if (sp) val = (val > 20.f) ? val : log1pf(expf(val));
                if (EPI == 1)
                    ((float*)Cout)[(size_t)row * ldc + col] = val;
                else
                    ((_Float16*)Cout)[(size_t)row * ldc + col] = (_Float16)val;
            }
        }
    }
}

// ---------------------------------------------------------------------------
// Time-conv (T channels, D2 spatial, k=3 pad=1) + SiLU; xp/xca f16 (M x D2P)
// ---------------------------------------------------------------------------
__global__ __launch_bounds__(256)
void conv_kernel(const _Float16* __restrict__ xp, const float* __restrict__ conv_w,
                 const float* __restrict__ conv_b, _Float16* __restrict__ xca) {
    __shared__ float w[432 + 12];
    for (int i = threadIdx.x; i < 432; i += 256) w[i] = conv_w[i];
    if (threadIdx.x < 12) w[432 + threadIdx.x] = conv_b[threadIdx.x];
    __syncthreads();

    int idx = blockIdx.x * 256 + threadIdx.x;
    if (idx >= BN_ROWS * D2P) return;
    int d   = idx % D2P;
    int bnr = idx / D2P;
    _Float16* outb = xca + (size_t)bnr * TT * D2P;
    if (d >= D2) {  // zero the padding columns
        for (int t = 0; t < TT; ++t) outb[t * D2P + d] = (_Float16)0.f;
        return;
    }
    const _Float16* base = xp + (size_t)bnr * TT * D2P;
    float xin[TT][3];
#pragma unroll
    for (int tin = 0; tin < TT; ++tin) {
        xin[tin][0] = (d > 0) ? (float)base[tin * D2P + d - 1] : 0.f;
        xin[tin][1] = (float)base[tin * D2P + d];
        xin[tin][2] = (float)base[tin * D2P + d + 1];  // d+1<=304: padded col is 0
    }
#pragma unroll
    for (int t = 0; t < TT; ++t) {
        float acc = w[432 + t];
#pragma unroll
        for (int tin = 0; tin < TT; ++tin) {
            acc += w[(t * TT + tin) * 3 + 0] * xin[tin][0];
            acc += w[(t * TT + tin) * 3 + 1] * xin[tin][1];
            acc += w[(t * TT + tin) * 3 + 2] * xin[tin][2];
        }
        float sv = acc / (1.f + expf(-acc));
        outb[t * D2P + d] = (_Float16)sv;
    }
}

// ---------------------------------------------------------------------------
// Combine (fused layout): fused[row] = [delta(0..319) | Bm(320..447) | Cm(448..575)]
// s = dot(Bm,Cm); xco <- silu(xco*delta*s) * silu(dpre)  (in place)
// ---------------------------------------------------------------------------
__global__ __launch_bounds__(256)
void combine_kernel(_Float16* __restrict__ xco, const _Float16* __restrict__ fused,
                    const _Float16* __restrict__ dpre) {
    int lane = threadIdx.x & 31, wid = threadIdx.x >> 5;
    int row = blockIdx.x * 8 + wid;
    if (row >= M_ROWS) return;
    const _Float16* fr = fused + (size_t)row * FUSEW;
    const _Float16* bp = fr + 320;
    const _Float16* cp = fr + 448;
    float s = 0.f;
    for (int i = lane; i < SSN; i += 32) s += (float)bp[i] * (float)cp[i];
    for (int off = 16; off; off >>= 1) s += __shfl_xor(s, off, 32);

    _Float16* xr = xco + (size_t)row * D2P;
    const _Float16* pr = dpre + (size_t)row * D2P;
    for (int c = lane; c < D2P; c += 32) {
        float y  = (float)xr[c] * (float)fr[c] * s;
        float a  = y / (1.f + expf(-y));
        float dp = (float)pr[c];
        float b2 = dp / (1.f + expf(-dp));
        xr[c] = (_Float16)(a * b2);
    }
}

// ---------------------------------------------------------------------------
// Head: out[b,j,n] = b_out[j] + sum_{t,c} h[bn, t, c] * W_out[t*152+c, j]
// ---------------------------------------------------------------------------
__global__ void head_kernel(const float* __restrict__ h, const float* __restrict__ W_out,
                            const float* __restrict__ b_out, float* __restrict__ out) {
    int idx = blockIdx.x * 256 + threadIdx.x;
    if (idx >= BQ * NN * 12) return;
    int j  = idx % 12;
    int bn = idx / 12;
    int b  = bn / NN;
    int n  = bn % NN;
    float acc = b_out[j];
    const float* hr = h + (size_t)bn * TT * DMP;
    for (int t = 0; t < TT; ++t)
        for (int c = 0; c < 152; ++c)
            acc += hr[t * DMP + c] * W_out[(t * 152 + c) * 12 + j];
    out[((size_t)b * 12 + j) * NN + n] = acc;
}

// ---------------------------------------------------------------------------
// Launch
// ---------------------------------------------------------------------------
extern "C" void kernel_launch(void* const* d_in, const int* in_sizes, int n_in,
                              void* d_out, int out_size, void* d_ws, size_t ws_size,
                              hipStream_t stream) {
    const float* x         = (const float*)d_in[0];
    const float* W_in      = (const float*)d_in[1];
    const float* b_in      = (const float*)d_in[2];
    const float* tod       = (const float*)d_in[3];
    const float* dow       = (const float*)d_in[4];
    const float* adp       = (const float*)d_in[5];
    const float* norm_w    = (const float*)d_in[6];
    const float* inproj_w  = (const float*)d_in[7];
    const float* inproj_b  = (const float*)d_in[8];
    const float* conv_w    = (const float*)d_in[9];
    const float* conv_b    = (const float*)d_in[10];
    const float* convlin_w = (const float*)d_in[11];
    const float* convlin_b = (const float*)d_in[12];
    const float* fc1_w     = (const float*)d_in[13];
    const float* fc1_b     = (const float*)d_in[14];
    const float* fc2_w     = (const float*)d_in[15];
    const float* fc2_b     = (const float*)d_in[16];
    const float* fc3_w     = (const float*)d_in[17];
    const float* fc3_b     = (const float*)d_in[18];
    // d_in[19] = A_ssm (unused: h0 = 0 => dA term vanishes)
    const float* D_w       = (const float*)d_in[20];
    const float* D_b       = (const float*)d_in[21];
    const float* outproj_w = (const float*)d_in[22];
    const float* outproj_b = (const float*)d_in[23];
    const float* W_out     = (const float*)d_in[24];
    const float* b_out     = (const float*)d_in[25];
    float* out = (float*)d_out;

    char* p = (char*)d_ws;
    auto alloc = [&](size_t bytes) -> void* {
        void* r = (void*)p;
        p += (bytes + 255) & ~(size_t)255;
        return r;
    };

    float*    hbuf  = (float*)    alloc((size_t)M_ROWS * DMP * 4);
    _Float16* xn    = (_Float16*) alloc((size_t)M_ROWS * DMP * 2);
    _Float16* xp    = (_Float16*) alloc((size_t)M_ROWS * D2P * 2);
    _Float16* xca   = (_Float16*) alloc((size_t)M_ROWS * D2P * 2);   // reused as dpre
    _Float16* xco   = (_Float16*) alloc((size_t)M_ROWS * D2P * 2);   // reused as x_comb
    _Float16* fused = (_Float16*) alloc((size_t)M_ROWS * FUSEW * 2); // delta|Bm|Cm

    _Float16 *Winp[3], *Wcvl[3], *Wfuse[3], *Wdw[3], *Wout[3];
    float    *Binp[3], *Bcvl[3], *Bfuse[3], *Bdw[3], *Bout[3];
    for (int i = 0; i < NLAYERS; ++i) {
        Winp[i]  = (_Float16*)alloc((size_t)320 * 160 * 2);   // [n][k], n<320, k<160
        Wcvl[i]  = (_Float16*)alloc((size_t)320 * 320 * 2);
        Wfuse[i] = (_Float16*)alloc((size_t)FUSEW * 320 * 2); // 576 x 320
        Wdw[i]   = (_Float16*)alloc((size_t)320 * 160 * 2);
        Wout[i]  = (_Float16*)alloc((size_t)192 * 320 * 2);
        Binp[i]  = (float*)alloc(320 * 4);
        Bcvl[i]  = (float*)alloc(320 * 4);
        Bfuse[i] = (float*)alloc(FUSEW * 4);
        Bdw[i]   = (float*)alloc(320 * 4);
        Bout[i]  = (float*)alloc(192 * 4);
    }

    auto padw = [&](const float* s, _Float16* d, int K, int Nn, int Kp, int Nseg, int col0) {
        pad_wT_f16<<<(Nseg * Kp + 255) / 256, 256, 0, stream>>>(s, d, K, Nn, Kp, Nseg, col0);
    };
    auto padb = [&](const float* s, float* d, int Nn, int Nseg, int col0) {
        pad_b_f32<<<(Nseg + 255) / 256, 256, 0, stream>>>(s, d, Nn, Nseg, col0);
    };
    for (int i = 0; i < NLAYERS; ++i) {
        padw(inproj_w  + (size_t)i * 152 * 304, Winp[i],  152, 304, 160, 320, 0);
        padw(convlin_w + (size_t)i * 304 * 304, Wcvl[i],  304, 304, 320, 320, 0);
        padw(fc1_w     + (size_t)i * 304 * 304, Wfuse[i], 304, 304, 320, 320, 0);
        padw(fc2_w     + (size_t)i * 304 * 128, Wfuse[i], 304, 128, 320, 128, 320);
        padw(fc3_w     + (size_t)i * 304 * 128, Wfuse[i], 304, 128, 320, 128, 448);
        padw(D_w       + (size_t)i * 152 * 304, Wdw[i],   152, 304, 160, 320, 0);
        padw(outproj_w + (size_t)i * 304 * 152, Wout[i],  304, 152, 320, 192, 0);
        padb(inproj_b  + (size_t)i * 304, Binp[i],  304, 320, 0);
        padb(convlin_b + (size_t)i * 304, Bcvl[i],  304, 320, 0);
        padb(fc1_b     + (size_t)i * 304, Bfuse[i], 304, 320, 0);
        padb(fc2_b     + (size_t)i * 128, Bfuse[i], 128, 128, 320);
        padb(fc3_b     + (size_t)i * 128, Bfuse[i], 128, 128, 448);
        padb(D_b       + (size_t)i * 304, Bdw[i],   304, 320, 0);
        padb(outproj_b + (size_t)i * 152, Bout[i],  152, 192, 0);
    }

    // Embedding -> h
    embed_kernel<<<(M_ROWS * DMP + 255) / 256, 256, 0, stream>>>(
        x, W_in, b_in, tod, dow, adp, hbuf);

    const int gmx = (M_ROWS + 127) / 128;        // 663
    const int rowsBlocks = (M_ROWS + 7) / 8;     // wave-per-row kernels

    for (int i = 0; i < NLAYERS; ++i) {
        rmsnorm_kernel<<<rowsBlocks, 256, 0, stream>>>(hbuf, norm_w + (size_t)i * 152, xn);

        // xp = xn @ inproj + b
        gemm_wmma<0><<<dim3(gmx, 5), 256, 0, stream>>>(
            xn, DMP, Winp[i], 160, Binp[i], (void*)xp, D2P, 320, 160, 0);

        // xca = silu(conv_time(xp))
        conv_kernel<<<(BN_ROWS * D2P + 255) / 256, 256, 0, stream>>>(
            xp, conv_w + (size_t)i * 432, conv_b + (size_t)i * 12, xca);

        // xco = xca @ convlin + b
        gemm_wmma<0><<<dim3(gmx, 5), 256, 0, stream>>>(
            xca, D2P, Wcvl[i], 320, Bcvl[i], (void*)xco, D2P, 320, 320, 0);

        // fused = [softplus(xco@fc1+b) | xco@fc2+b | xco@fc3+b]  (one pass over xco)
        gemm_wmma<0><<<dim3(gmx, 9), 256, 0, stream>>>(
            xco, D2P, Wfuse[i], 320, Bfuse[i], (void*)fused, FUSEW, FUSEW, 320, 320);

        // dpre = xn @ D_w + b   (into xca buffer)
        gemm_wmma<0><<<dim3(gmx, 5), 256, 0, stream>>>(
            xn, DMP, Wdw[i], 160, Bdw[i], (void*)xca, D2P, 320, 160, 0);

        // xco <- silu(xco*delta*s) * silu(dpre)  in place
        combine_kernel<<<rowsBlocks, 256, 0, stream>>>(xco, fused, xca);

        // h = xco @ outproj + b   (f32, padded cols become 0)
        gemm_wmma<1><<<dim3(gmx, 3), 256, 0, stream>>>(
            xco, D2P, Wout[i], 320, Bout[i], (void*)hbuf, DMP, DMP, 320, 0);
    }

    head_kernel<<<(BQ * NN * 12 + 255) / 256, 256, 0, stream>>>(hbuf, W_out, b_out, out);

    (void)in_sizes; (void)n_in; (void)out_size; (void)ws_size;
}
